// ModulatedChunks_61229053772535
// MI455X (gfx1250) — matrix operand order repaired and checked by reading, not compiled, not run on hardware
//
#include <hip/hip_runtime.h>
#include <hip/hip_bf16.h>

// ---------------------------------------------------------------------------
// ModulatedChunks for MI455X (gfx1250, wave32, WMMA + Tensor Data Mover).
// Heavy GEMMs run in bf16 via v_wmma_f32_16x16x32_bf16 with f32 accumulate.
// Global->LDS tile staging is done by TDM (tensor_load_to_lds) issued by one
// wave per block, synchronized with s_wait_tensorcnt + workgroup barriers.
// ---------------------------------------------------------------------------

typedef __bf16 bf16_t;
typedef bf16_t v16bf __attribute__((ext_vector_type(16)));
typedef float  v8f   __attribute__((ext_vector_type(8)));

typedef unsigned int u32x4 __attribute__((ext_vector_type(4)));
typedef int          i32x4 __attribute__((ext_vector_type(4)));
typedef int          i32x8 __attribute__((ext_vector_type(8)));

union Frag {
    v16bf v;
    struct { uint4 lo; uint4 hi; } q;
};

static constexpr int B_  = 4;
static constexpr int T_  = 1024;
static constexpr int C_  = 1024;
static constexpr int QD_ = 1024;
static constexpr int NW_ = 32;
static constexpr int CS_ = 4;              // ws/nc == 4 for every scale
static constexpr int S_  = T_ - CS_ + 1;   // 1021 sliding chunks

// low 32 bits of a flat shared-aperture pointer == wave-relative LDS address
// (ISA 10.2)
__device__ __forceinline__ uint32_t lds_lo32(const void* p)
{
    return (uint32_t)(uint64_t)(uintptr_t)p;
}

// ---------------------------------------------------------------------------
// TDM: DMA a [rows x 32] bf16 tile (row stride Kelems) from global memory
// into LDS with a 16B pad after every 64B row (=> 80B LDS row stride).
// Issued by a single wave; tracked with TENSORcnt.
// ---------------------------------------------------------------------------
__device__ __forceinline__ void tdm_load_tile(const bf16_t* gptr, uint32_t lds_addr,
                                              int rows, int Kelems)
{
    uint64_t ga = (uint64_t)(uintptr_t)gptr;

    // ---- D# group 0: count=1, lds_addr, 57b global_addr, type=2 (image)
    u32x4 g0;
    g0.x = 1u;                                       // count = 1 valid descriptor
    g0.y = lds_addr;                                 // LDS byte address
    g0.z = (uint32_t)ga;                             // global_addr[31:0]
    g0.w = (uint32_t)((ga >> 32) & 0x1FFFFFFu)       // global_addr[56:32]
         | (2u << 30);                               // type = 2

    // ---- D# group 1 bitfields (256 bits)
    // [17:16] data_size=1 (2B)  [20] pad_enable  [24:22] pad_interval=3 (16 DW)
    // [31:25] pad_amount=3 (4 DW)  [79:48] tensor_dim0  [111:80] tensor_dim1
    // [127:112] tile_dim0=32  [143:128] tile_dim1=rows  [207:160] dim0 stride
    uint64_t q0 = (uint64_t)((1u << 16) | (1u << 20) | (3u << 22) | (3u << 25))
                | ((uint64_t)((uint32_t)Kelems & 0xFFFFu) << 48);
    uint64_t q1 = (uint64_t)((uint32_t)Kelems >> 16)
                | ((uint64_t)(uint32_t)rows << 16)
                | ((uint64_t)32u << 48);
    uint64_t q2 = (uint64_t)((uint32_t)rows & 0xFFFFu)
                | ((uint64_t)(uint32_t)Kelems << 32);
    uint64_t q3 = 0;

    union { uint64_t q[4]; i32x8 v; } g1u;
    g1u.q[0] = q0; g1u.q[1] = q1; g1u.q[2] = q2; g1u.q[3] = q3;

    i32x4 z4 = {0, 0, 0, 0};
#if defined(__clang_major__) && (__clang_major__ >= 23)
    i32x8 z8 = {0, 0, 0, 0, 0, 0, 0, 0};
    __builtin_amdgcn_tensor_load_to_lds(g0, g1u.v, z4, z4, z8, 0);
#else
    __builtin_amdgcn_tensor_load_to_lds(g0, g1u.v, z4, z4, 0);
#endif
}

// ---------------------------------------------------------------------------
// Positional-encoding add + f32 -> bf16 convert.
// ---------------------------------------------------------------------------
__global__ __launch_bounds__(256)
void pe_cvt_kernel(const float* __restrict__ src, bf16_t* __restrict__ dst,
                   int total, int C, int period)
{
    int idx = blockIdx.x * blockDim.x + threadIdx.x;
    if (idx >= total) return;
    int c = idx % C;
    int t = (idx / C) % period;
    float i2  = (float)(c & ~1);
    float ang = (float)t * __expf(i2 * (-9.210340371976184f / (float)C));
    float pe  = (c & 1) ? __cosf(ang) : __sinf(ang);
    dst[idx] = (bf16_t)(src[idx] + pe);
}

__global__ __launch_bounds__(256)
void cvt_bf16_kernel(const float* __restrict__ src, bf16_t* __restrict__ dst, int n)
{
    int idx = blockIdx.x * blockDim.x + threadIdx.x;
    if (idx < n) dst[idx] = (bf16_t)src[idx];
}

// ---------------------------------------------------------------------------
// WMMA GEMM:  D[m,n] = act( sum_k A[m,k] * W[n,k] + bias[n] )
// A: M x K bf16 row-major, W: N x K bf16 row-major.
// TDM stages double-buffered LDS tiles (80B padded rows -> conflict-free b128
// fragment reads); 8 wave32 waves run pure WMMA.
// grid = (N/BN, M/BM, batch)
// ---------------------------------------------------------------------------
template<int BM, int BN, int WM, int WN, bool RELU, bool BIAS, bool WF32, bool WBF16>
__global__ __launch_bounds__((BM / WM) * (BN / WN) * 32)
void wmma_gemm_kernel(const bf16_t* __restrict__ A, long strideA,
                      const bf16_t* __restrict__ W, long strideW,
                      const float*  __restrict__ bias,
                      float*  __restrict__ Cf,
                      bf16_t* __restrict__ Cb, long strideC,
                      int M, int N, int K)
{
    constexpr int BK      = 32;
    constexpr int LDK     = BK + 8;      // 80B rows -> no LDS bank conflicts
    constexpr int WAVES_M = BM / WM;
    constexpr int MI      = WM / 16;
    constexpr int NI      = WN / 16;

    __shared__ __align__(16) bf16_t As[2][BM][LDK];
    __shared__ __align__(16) bf16_t Bs[2][BN][LDK];

    const int tid  = threadIdx.x;
    const int wave = tid >> 5;
    const int lane = tid & 31;
    const int wm   = wave % WAVES_M;
    const int wn   = wave / WAVES_M;

    const int bm0 = blockIdx.y * BM;
    const int bn0 = blockIdx.x * BN;
    const int bat = blockIdx.z;

    const bf16_t* Ab = A + (long)bat * strideA + (long)bm0 * K;
    const bf16_t* Wb = W + (long)bat * strideW + (long)bn0 * K;

    v8f acc[MI][NI];
#pragma unroll
    for (int i = 0; i < MI; ++i)
#pragma unroll
        for (int j = 0; j < NI; ++j)
#pragma unroll
            for (int r = 0; r < 8; ++r) acc[i][j][r] = 0.0f;

    // wave 0 drives the Tensor Data Mover (TDM ignores EXEC -> keep the issue
    // in a wave-uniform branch so exactly one DMA per tile is launched)
    if (wave == 0) {
        tdm_load_tile(Ab, lds_lo32(&As[0][0][0]), BM, K);
        tdm_load_tile(Wb, lds_lo32(&Bs[0][0][0]), BN, K);
    }

    int buf = 0;
    const int half = lane >> 4;
    const int l16  = lane & 15;

    for (int k0 = 0; k0 < K; k0 += BK) {
        // waves that issued TDM wait for LDS fill; others pass instantly
        __builtin_amdgcn_s_wait_tensorcnt(0);
        __syncthreads();

        if (k0 + BK < K && wave == 0) {
            tdm_load_tile(Ab + (k0 + BK), lds_lo32(&As[buf ^ 1][0][0]), BM, K);
            tdm_load_tile(Wb + (k0 + BK), lds_lo32(&Bs[buf ^ 1][0][0]), BN, K);
        }

        Frag af[MI], bf[NI];
        // A 16x32 bf16 frag: lanes 0-15 hold K[0:8)+K[16:24), lanes 16-31
        // hold K[8:16)+K[24:32)  (ISA 7.12.2), two 16B LDS reads per lane.
#pragma unroll
        for (int i = 0; i < MI; ++i) {
            const bf16_t* p = &As[buf][wm * WM + i * 16 + l16][half * 8];
            af[i].q.lo = *(const uint4*)p;
            af[i].q.hi = *(const uint4*)(p + 16);
        }
        // B 32x16 bf16 frag: lanes 0-15 hold K[0:8)+K[16:24), lanes 16-31 K[8:16)+K[24:32).
#pragma unroll
        for (int j = 0; j < NI; ++j) {
            const bf16_t* p = &Bs[buf][wn * WN + j * 16 + l16][half * 8];
            bf[j].q.lo = *(const uint4*)p;
            bf[j].q.hi = *(const uint4*)(p + 16);
        }
#pragma unroll
        for (int i = 0; i < MI; ++i)
#pragma unroll
            for (int j = 0; j < NI; ++j)
                acc[i][j] = __builtin_amdgcn_wmma_f32_16x16x32_bf16(
                    false, af[i].v, false, bf[j].v,
                    (short)0, acc[i][j], false, false);

        buf ^= 1;
    }

    // Epilogue: C/D layout => VGPR r holds row r (lanes 0-15) / row r+8
    // (lanes 16-31), column = lane & 15.
    const int rh = (lane >> 4) * 8;
#pragma unroll
    for (int j = 0; j < NI; ++j) {
        int n = bn0 + wn * WN + j * 16 + l16;
        float bv = BIAS ? bias[n] : 0.0f;
#pragma unroll
        for (int i = 0; i < MI; ++i) {
            int m0 = bm0 + wm * WM + i * 16 + rh;
#pragma unroll
            for (int r = 0; r < 8; ++r) {
                float v = acc[i][j][r] + bv;
                if (RELU) v = fmaxf(v, 0.0f);
                long off = (long)bat * strideC + (long)(m0 + r) * N + n;
                if (WF32)  Cf[off] = v;
                if (WBF16) Cb[off] = (bf16_t)v;
            }
        }
    }
}

// ---------------------------------------------------------------------------
// Per-chunk kernel. One wave per (b, s), s in [0, 1021):
//  lane w: 4-sample sliding sum of sim -> butterfly argmax (first-index ties)
//  then 4 fused dot products over C with enc2[b,label] * mean4(vis).
// ---------------------------------------------------------------------------
__global__ __launch_bounds__(256)
void chunk_kernel(const float* __restrict__ vis2, const float* __restrict__ sim,
                  const float* __restrict__ enc2,
                  const float* __restrict__ seg_w, const float* __restrict__ lp_w,
                  const float* __restrict__ rp_w,
                  float* __restrict__ chunkvals)
{
    int widx = (blockIdx.x * blockDim.x + threadIdx.x) >> 5;
    int lane = threadIdx.x & 31;
    if (widx >= B_ * S_) return;
    int b = widx / S_;
    int s = widx % S_;

    // sliding-4 sum of sim over t, one NW column per lane (NW_ == 32)
    const float* sp = sim + ((long)b * T_ + s) * NW_ + lane;
    float sv = sp[0] + sp[NW_] + sp[2 * NW_] + sp[3 * NW_];
    int bi = lane;
#pragma unroll
    for (int off = 16; off; off >>= 1) {
        float ov = __shfl_xor(sv, off, 32);
        int   oi = __shfl_xor(bi, off, 32);
        if (ov > sv || (ov == sv && oi < bi)) { sv = ov; bi = oi; }
    }
    const int lab = bi;  // identical in every lane after full butterfly

    const float* vp = vis2 + ((long)b * T_ + s) * C_;
    const float* ep = enc2 + ((long)b * NW_ + lab) * C_;
    float a0 = 0.f, a1 = 0.f, a2 = 0.f, a3 = 0.f;
    for (int c = lane; c < C_; c += 32) {
        float mean = 0.25f * (vp[c] + vp[c + C_] + vp[c + 2 * C_] + vp[c + 3 * C_]);
        float m = ep[c] * mean;
        a0 = fmaf(m, seg_w[c],       a0);
        a1 = fmaf(m, seg_w[C_ + c],  a1);
        a2 = fmaf(m, lp_w[c],        a2);
        a3 = fmaf(m, rp_w[c],        a3);
    }
#pragma unroll
    for (int off = 16; off; off >>= 1) {
        a0 += __shfl_xor(a0, off, 32);
        a1 += __shfl_xor(a1, off, 32);
        a2 += __shfl_xor(a2, off, 32);
        a3 += __shfl_xor(a3, off, 32);
    }
    if (lane == 0)
        *(float4*)(chunkvals + (long)widx * 4) = make_float4(a0, a1, a2, a3);
}

// ---------------------------------------------------------------------------
// Expand per-chunk values into one scale's seg/lp/rp outputs (+ biases).
// seg: (B, nc, L, 2); lp, rp: (B, L, nc).
// ---------------------------------------------------------------------------
__global__ __launch_bounds__(256)
void scatter_kernel(const float* __restrict__ chunkvals,
                    const float* __restrict__ seg_b,
                    const float* __restrict__ lp_b,
                    const float* __restrict__ rp_b,
                    float* __restrict__ out_seg,
                    float* __restrict__ out_lp,
                    float* __restrict__ out_rp,
                    int L, int nc)
{
    int idx = blockIdx.x * blockDim.x + threadIdx.x;
    int total = B_ * L * nc;
    if (idx >= total) return;
    int j = idx % nc;
    int l = (idx / nc) % L;
    int b = idx / (nc * L);
    int s = l + j * CS_;
    const float* v = chunkvals + ((long)b * S_ + s) * 4;
    long segoff = (((long)b * nc + j) * L + l) * 2;
    out_seg[segoff + 0] = v[0] + seg_b[0];
    out_seg[segoff + 1] = v[1] + seg_b[1];
    out_lp[((long)b * L + l) * nc + j] = v[2] + lp_b[0];
    out_rp[((long)b * L + l) * nc + j] = v[3] + rp_b[0];
}

// ---------------------------------------------------------------------------

extern "C" void kernel_launch(void* const* d_in, const int* in_sizes, int n_in,
                              void* d_out, int out_size, void* d_ws, size_t ws_size,
                              hipStream_t stream)
{
    (void)in_sizes; (void)n_in; (void)out_size; (void)ws_size;

    const float* vis_feats = (const float*)d_in[0];
    const float* query     = (const float*)d_in[1];
    const float* ve1_w1    = (const float*)d_in[2];
    const float* ve1_b1    = (const float*)d_in[3];
    const float* ve1_w2    = (const float*)d_in[4];
    const float* ve1_b2    = (const float*)d_in[5];
    const float* se1_w1    = (const float*)d_in[6];
    const float* se1_b1    = (const float*)d_in[7];
    const float* se1_w2    = (const float*)d_in[8];
    const float* se1_b2    = (const float*)d_in[9];
    const float* se2_w1    = (const float*)d_in[10];
    const float* se2_b1    = (const float*)d_in[11];
    const float* se2_w2    = (const float*)d_in[12];
    const float* se2_b2    = (const float*)d_in[13];
    const float* seg_w     = (const float*)d_in[14];
    const float* seg_b     = (const float*)d_in[15];
    const float* lp_w      = (const float*)d_in[16];
    const float* lp_b      = (const float*)d_in[17];
    const float* rp_w      = (const float*)d_in[18];
    const float* rp_b      = (const float*)d_in[19];

    // -------- workspace carve-out (~56 MB) --------
    char* ws = (char*)d_ws;
    size_t off = 0;
    auto carve = [&](size_t bytes) {
        void* p = ws + off;
        off += (bytes + 255) & ~(size_t)255;
        return p;
    };
    bf16_t* vis_pe = (bf16_t*)carve((size_t)B_ * T_ * C_ * 2);
    bf16_t* h1     = (bf16_t*)carve((size_t)B_ * T_ * C_ * 2);
    float*  vis2f  = (float*) carve((size_t)B_ * T_ * C_ * 4);
    bf16_t* vis2b  = (bf16_t*)carve((size_t)B_ * T_ * C_ * 2);
    bf16_t* q_pe   = (bf16_t*)carve((size_t)B_ * NW_ * QD_ * 2);
    bf16_t* henc   = (bf16_t*)carve((size_t)B_ * NW_ * QD_ * 2);
    bf16_t* enc1b  = (bf16_t*)carve((size_t)B_ * NW_ * C_ * 2);
    float*  enc2f  = (float*) carve((size_t)B_ * NW_ * C_ * 4);
    bf16_t* wv1b   = (bf16_t*)carve((size_t)C_ * C_ * 2);
    bf16_t* wv2b   = (bf16_t*)carve((size_t)C_ * C_ * 2);
    bf16_t* ws11b  = (bf16_t*)carve((size_t)QD_ * QD_ * 2);
    bf16_t* ws12b  = (bf16_t*)carve((size_t)C_ * QD_ * 2);
    bf16_t* ws21b  = (bf16_t*)carve((size_t)QD_ * QD_ * 2);
    bf16_t* ws22b  = (bf16_t*)carve((size_t)C_ * QD_ * 2);
    float*  simf   = (float*) carve((size_t)B_ * T_ * NW_ * 4);
    float*  chunkvals = (float*)carve((size_t)B_ * S_ * 4 * 4);

    // -------- stage 1: PE add + bf16 convert --------
    {
        int total = B_ * T_ * C_;
        pe_cvt_kernel<<<(total + 255) / 256, 256, 0, stream>>>(
            vis_feats, vis_pe, total, C_, T_);
        total = B_ * NW_ * QD_;
        pe_cvt_kernel<<<(total + 255) / 256, 256, 0, stream>>>(
            query, q_pe, total, QD_, NW_);
    }
    {
        const float* srcs[6] = {ve1_w1, ve1_w2, se1_w1, se1_w2, se2_w1, se2_w2};
        bf16_t* dsts[6]      = {wv1b, wv2b, ws11b, ws12b, ws21b, ws22b};
        for (int i = 0; i < 6; ++i) {
            int n = 1024 * 1024;
            cvt_bf16_kernel<<<(n + 255) / 256, 256, 0, stream>>>(srcs[i], dsts[i], n);
        }
    }

    // -------- stage 2: GEMMs (bf16 WMMA + TDM staging) --------
    {
        dim3 blk(256);
        // h1 = relu(vis_pe @ ve1_w1^T + b1)   [4096 x 1024 x 1024]
        wmma_gemm_kernel<128,128,32,64, true,  true,  false, true>
            <<<dim3(1024/128, 4096/128, 1), blk, 0, stream>>>(
                vis_pe, 0, wv1b, 0, ve1_b1, nullptr, h1, 0, 4096, 1024, 1024);
        // vis2 = h1 @ ve1_w2^T + b2  -> f32 (chunk means) + bf16 (sim GEMM)
        wmma_gemm_kernel<128,128,32,64, false, true,  true,  true>
            <<<dim3(1024/128, 4096/128, 1), blk, 0, stream>>>(
                h1, 0, wv2b, 0, ve1_b2, vis2f, vis2b, 0, 4096, 1024, 1024);
        // henc = relu(q_pe @ se1_w1^T + b1)   [128 x 1024 x 1024]
        wmma_gemm_kernel<128,128,32,64, true,  true,  false, true>
            <<<dim3(1024/128, 1, 1), blk, 0, stream>>>(
                q_pe, 0, ws11b, 0, se1_b1, nullptr, henc, 0, 128, 1024, 1024);
        // enc1 = henc @ se1_w2^T + b2 (bf16, consumed by sim GEMM)
        wmma_gemm_kernel<128,128,32,64, false, true,  false, true>
            <<<dim3(1024/128, 1, 1), blk, 0, stream>>>(
                henc, 0, ws12b, 0, se1_b2, nullptr, enc1b, 0, 128, 1024, 1024);
        // henc = relu(q_pe @ se2_w1^T + b1)  (reuse buffer; stream-ordered)
        wmma_gemm_kernel<128,128,32,64, true,  true,  false, true>
            <<<dim3(1024/128, 1, 1), blk, 0, stream>>>(
                q_pe, 0, ws21b, 0, se2_b1, nullptr, henc, 0, 128, 1024, 1024);
        // enc2 = henc @ se2_w2^T + b2 (f32, consumed by chunk kernel)
        wmma_gemm_kernel<128,128,32,64, false, true,  true,  false>
            <<<dim3(1024/128, 1, 1), blk, 0, stream>>>(
                henc, 0, ws22b, 0, se2_b2, enc2f, nullptr, 0, 128, 1024, 1024);
        // sim[b] = vis2[b] @ enc1[b]^T   [batched 1024 x 32 x 1024]
        wmma_gemm_kernel<128,32,32,16, false, false, true,  false>
            <<<dim3(32/32, 1024/128, B_), blk, 0, stream>>>(
                vis2b, (long)T_ * C_, enc1b, (long)NW_ * C_, nullptr,
                simf, nullptr, (long)T_ * NW_, 1024, 32, 1024);
    }

    // -------- stage 3: per-chunk argmax + fused dot products --------
    {
        int waves  = B_ * S_;
        int blocks = (waves * 32 + 255) / 256;
        chunk_kernel<<<blocks, 256, 0, stream>>>(
            vis2f, simf, enc2f, seg_w, lp_w, rp_w, chunkvals);
    }

    // -------- stage 4: scatter into the 9 output tensors --------
    {
        float* outp = (float*)d_out;
        size_t o = 0;
        const int Ls[3]  = {1009, 993, 961};
        const int ncs[3] = {4, 8, 16};
        for (int si = 0; si < 3; ++si) {
            int L = Ls[si], nc = ncs[si];
            float* oseg = outp + o; o += (size_t)B_ * nc * L * 2;
            float* olp  = outp + o; o += (size_t)B_ * L * nc;
            float* orp  = outp + o; o += (size_t)B_ * L * nc;
            int total = B_ * L * nc;
            scatter_kernel<<<(total + 255) / 256, 256, 0, stream>>>(
                chunkvals, seg_b, lp_b, rp_b, oseg, olp, orp, L, nc);
        }
    }
}
